// GraphPolicyValueNetwork_83940840833580
// MI455X (gfx1250) — matrix operand order repaired and checked by reading, not compiled, not run on hardware
//
#include <hip/hip_runtime.h>
#include <math.h>

// ---------------------------------------------------------------------------
// CDNA5 WMMA types (wave32): v_wmma_f32_16x16x32_bf16
// ---------------------------------------------------------------------------
typedef __attribute__((ext_vector_type(16))) __bf16 v16bf;
typedef __attribute__((ext_vector_type(8)))  __bf16 bf16x8;
typedef __attribute__((ext_vector_type(8)))  float  v8f;

__device__ __forceinline__ __bf16 f2bf(float f) {
    union { float f; unsigned int u; } c; c.f = f;
    unsigned int r = c.u + 0x7FFFu + ((c.u >> 16) & 1u);   // RNE
    unsigned short s = (unsigned short)(r >> 16);
    return __builtin_bit_cast(__bf16, s);
}
__device__ __forceinline__ float bf2f(__bf16 b) {
    unsigned short s = __builtin_bit_cast(unsigned short, b);
    unsigned int u = (unsigned int)s << 16;
    return __builtin_bit_cast(float, u);
}

#define WMMA_BF16(AF, BF, ACC) \
    __builtin_amdgcn_wmma_f32_16x16x32_bf16(false, (AF), false, (BF), (short)0, (ACC), false, false)

// ---------------------------------------------------------------------------
// Weight packing: W[K,N] f32 -> bf16 fragments, group-interleaved:
//   Bp[(((group*KT + kt)*4 + t)*32 + lane)*16 + e]
//   group = tile/4, t = tile%4
//   element e of lane L <-> K = kt*32 + 8*(L>>4) + (e&7) + 16*(e>>3), col = tile*16 + (L&15)
// All 4 fragments of a (group,kt) are contiguous (4KB) -> immediate offsets
// in the GEMM inner loop. Ragged N zero-padded here.
// ---------------------------------------------------------------------------
__global__ void pack_weight_kernel(const float* __restrict__ W, __bf16* __restrict__ Bp,
                                   int K, int N, int nTilesPad)
{
    int idx = blockIdx.x * blockDim.x + threadIdx.x;
    int KT = K >> 5;
    int total = nTilesPad * KT * 32;
    if (idx >= total) return;
    int lane = idx & 31;
    int kt   = (idx >> 5) % KT;
    int tile = (idx >> 5) / KT;
    int grp  = tile >> 2;
    int t    = tile & 3;
    int hi   = lane >> 4;
    int col  = tile * 16 + (lane & 15);
    size_t outIdx = ((((size_t)grp * KT + kt) * 4 + t) * 32 + lane) * 16;
    __bf16* out = Bp + outIdx;
#pragma unroll
    for (int e = 0; e < 16; ++e) {
        int kk = kt * 32 + 8 * hi + (e & 7) + 16 * (e >> 3); // always < K (K % 32 == 0)
        float v = (col < N) ? W[(size_t)kk * N + col] : 0.f;
        out[e] = f2bf(v);
    }
}

__global__ void f32_to_bf16_kernel(const float* __restrict__ in, __bf16* __restrict__ out,
                                   long long n)
{
    long long i = (long long)blockIdx.x * blockDim.x + threadIdx.x;
    if (i < n) out[i] = f2bf(in[i]);
}

// ---------------------------------------------------------------------------
// GEMM: C[M,N] = act( A[M,K](bf16) @ W[K,N](packed bf16) (+ bias) )
// M % 16 == 0, K % 32 == 0 guaranteed by launcher.
// Each wave: 16x64 strip (4 accumulators). Software-pipelined: fragments for
// step kt+1 are loaded before the 4 WMMAs of step kt, so v_wmma overlaps the
// global_load_b128 latency instead of s_wait_loadcnt 0 on every WMMA.
// blockDim = (32, 4). grid = (M/16, ceil(nGroups/4)).
// ---------------------------------------------------------------------------
template <bool RELU, bool BIAS, bool OUTBF>
__global__ void wmma_gemm_kernel(const __bf16* __restrict__ A,
                                 const __bf16* __restrict__ Bp,
                                 const float* __restrict__ bias,
                                 void* __restrict__ Cout,
                                 int M, int N, int K, int nGroups)
{
    constexpr int TN = 4;
    const int lane  = threadIdx.x;
    const int group = blockIdx.y * blockDim.y + threadIdx.y;
    if (group >= nGroups) return;                  // wave-uniform exit
    const int mBase = blockIdx.x * 16;
    const int hi    = lane >> 4;
    const int row   = mBase + (lane & 15);
    const int KT    = K >> 5;

    v8f zero = {0.f, 0.f, 0.f, 0.f, 0.f, 0.f, 0.f, 0.f};
    v8f acc0 = zero, acc1 = zero, acc2 = zero, acc3 = zero;

    const __bf16* aPtr = A + (size_t)row * K + hi * 8;
    const __bf16* bPtr = Bp + ((size_t)group * KT * TN * 32 + lane) * 16;
    // per k-step: A advances 32 elems; B advances TN*32*16 = 2048 elems (4KB)

    // ---- preload kt = 0 ----
    bf16x8 alo = *(const bf16x8*)(aPtr);
    bf16x8 ahi = *(const bf16x8*)(aPtr + 16);
    v16bf af = __builtin_shufflevector(alo, ahi,
        0, 1, 2, 3, 4, 5, 6, 7, 8, 9, 10, 11, 12, 13, 14, 15);
    v16bf b0 = *(const v16bf*)(bPtr + 0 * 512);
    v16bf b1 = *(const v16bf*)(bPtr + 1 * 512);
    v16bf b2 = *(const v16bf*)(bPtr + 2 * 512);
    v16bf b3 = *(const v16bf*)(bPtr + 3 * 512);

    for (int kt = 0; kt < KT - 1; ++kt) {
        const __bf16* aN = aPtr + (size_t)(kt + 1) * 32;
        const __bf16* bN = bPtr + (size_t)(kt + 1) * (TN * 32 * 16);
        // issue next-step loads first (overlap with WMMAs below)
        bf16x8 nlo = *(const bf16x8*)(aN);
        bf16x8 nhi = *(const bf16x8*)(aN + 16);
        v16bf nb0 = *(const v16bf*)(bN + 0 * 512);
        v16bf nb1 = *(const v16bf*)(bN + 1 * 512);
        v16bf nb2 = *(const v16bf*)(bN + 2 * 512);
        v16bf nb3 = *(const v16bf*)(bN + 3 * 512);
        // compute current step
        acc0 = WMMA_BF16(af, b0, acc0);
        acc1 = WMMA_BF16(af, b1, acc1);
        acc2 = WMMA_BF16(af, b2, acc2);
        acc3 = WMMA_BF16(af, b3, acc3);
        af = __builtin_shufflevector(nlo, nhi,
            0, 1, 2, 3, 4, 5, 6, 7, 8, 9, 10, 11, 12, 13, 14, 15);
        b0 = nb0; b1 = nb1; b2 = nb2; b3 = nb3;
    }
    acc0 = WMMA_BF16(af, b0, acc0);
    acc1 = WMMA_BF16(af, b1, acc1);
    acc2 = WMMA_BF16(af, b2, acc2);
    acc3 = WMMA_BF16(af, b3, acc3);

    // ---- store: VGPR vv -> row mBase + vv + 8*hi; col = tile*16 + (lane&15)
    v8f accs[TN] = {acc0, acc1, acc2, acc3};
#pragma unroll
    for (int t = 0; t < TN; ++t) {
        int col = (group * TN + t) * 16 + (lane & 15);
        if (col < N) {
            float bv = BIAS ? bias[col] : 0.f;
#pragma unroll
            for (int vv = 0; vv < 8; ++vv) {
                int r = mBase + vv + hi * 8;
                float val = accs[t][vv] + bv;
                if (RELU) val = fmaxf(val, 0.f);
                if (OUTBF) ((__bf16*)Cout)[(size_t)r * N + col] = f2bf(val);
                else       ((float*)Cout)[(size_t)r * N + col] = val;
            }
        }
    }
}

// ---------------------------------------------------------------------------
// Degree / normalization
// ---------------------------------------------------------------------------
__global__ void degree_kernel(const int* __restrict__ dst, float* __restrict__ deg, int nEdges)
{
    int i = blockIdx.x * blockDim.x + threadIdx.x;
    if (i < nEdges) atomicAdd(&deg[dst[i]], 1.0f);
}

__global__ void rsqrt_kernel(const float* __restrict__ deg, float* __restrict__ dis, int n)
{
    int i = blockIdx.x * blockDim.x + threadIdx.x;
    if (i < n) dis[i] = rsqrtf(deg[i] + 1.0f);           // +1 self-loop
}

// ---------------------------------------------------------------------------
// Edge scatter-add: agg[dst] += hw[src] * dis[src]*dis[dst]  (float4 chunks)
// ---------------------------------------------------------------------------
__global__ void edge_aggregate_kernel(const float* __restrict__ hw,
                                      const int* __restrict__ src,
                                      const int* __restrict__ dst,
                                      const float* __restrict__ dis,
                                      float* __restrict__ agg,
                                      int nEdges, int H)
{
    const int chunks = H >> 2;
    const long long total  = (long long)nEdges * chunks;
    const long long stride = (long long)gridDim.x * blockDim.x;
    for (long long idx = (long long)blockIdx.x * blockDim.x + threadIdx.x;
         idx < total; idx += stride) {
        int e = (int)(idx / chunks);
        int c = (int)(idx % chunks) << 2;
        int s = src[e], d = dst[e];
        float norm = dis[s] * dis[d];
        const float4 hv = *(const float4*)(hw + (size_t)s * H + c);
        float* out = agg + (size_t)d * H + c;
        atomicAdd(out + 0, hv.x * norm);
        atomicAdd(out + 1, hv.y * norm);
        atomicAdd(out + 2, hv.z * norm);
        atomicAdd(out + 3, hv.w * norm);
    }
}

// h_next(bf16) = relu(agg + hw * dis^2 + bias)
__global__ void selfloop_bias_relu_kernel(const float* __restrict__ agg,
                                          const float* __restrict__ hw,
                                          const float* __restrict__ dis,
                                          const float* __restrict__ bias,
                                          __bf16* __restrict__ out,
                                          int n, int H)
{
    long long idx = (long long)blockIdx.x * blockDim.x + threadIdx.x;
    long long total = (long long)n * H;
    if (idx < total) {
        int i = (int)(idx / H);
        int j = (int)(idx % H);
        float d = dis[i];
        float v = agg[idx] + hw[idx] * d * d + bias[j];
        out[idx] = f2bf(fmaxf(v, 0.f));
    }
}

// ---------------------------------------------------------------------------
// Global mean pool (h in bf16)
// ---------------------------------------------------------------------------
__global__ void pool_accum_kernel(const __bf16* __restrict__ h,
                                  const int* __restrict__ batch,
                                  float* __restrict__ sums,
                                  float* __restrict__ cnts,
                                  int n, int H)
{
    const int chunks = H >> 2;
    long long idx = (long long)blockIdx.x * blockDim.x + threadIdx.x;
    long long total = (long long)n * chunks;
    if (idx < total) {
        int node = (int)(idx / chunks);
        int c    = (int)(idx % chunks) << 2;
        int g    = batch[node];
        const __bf16* hp = h + (size_t)node * H + c;
        float* out = sums + (size_t)g * H + c;
        atomicAdd(out + 0, bf2f(hp[0]));
        atomicAdd(out + 1, bf2f(hp[1]));
        atomicAdd(out + 2, bf2f(hp[2]));
        atomicAdd(out + 3, bf2f(hp[3]));
        if (c == 0) atomicAdd(&cnts[g], 1.0f);
    }
}

__global__ void pool_div_kernel(const float* __restrict__ sums,
                                const float* __restrict__ cnts,
                                __bf16* __restrict__ g,
                                int G, int H)
{
    int idx = blockIdx.x * blockDim.x + threadIdx.x;
    if (idx < G * H) g[idx] = f2bf(sums[idx] / fmaxf(cnts[idx / H], 1.0f));
}

// ---------------------------------------------------------------------------
// Softmax over C columns, one 256-thread block per row
// ---------------------------------------------------------------------------
__global__ void softmax_kernel(const float* __restrict__ logits,
                               float* __restrict__ out, int C)
{
    __shared__ float red[256];
    int g = blockIdx.x;
    int t = threadIdx.x;
    float v = (t < C) ? logits[(size_t)g * C + t] : -INFINITY;
    red[t] = v;
    __syncthreads();
    for (int s = 128; s > 0; s >>= 1) {
        if (t < s) red[t] = fmaxf(red[t], red[t + s]);
        __syncthreads();
    }
    float mx = red[0];
    __syncthreads();
    float e = (t < C) ? __expf(v - mx) : 0.f;
    red[t] = e;
    __syncthreads();
    for (int s = 128; s > 0; s >>= 1) {
        if (t < s) red[t] += red[t + s];
        __syncthreads();
    }
    float sum = red[0];
    if (t < C) out[(size_t)g * C + t] = e / sum;
}

// value = tanh(v1 . vw2 + vb2)
__global__ void value_kernel(const float* __restrict__ v1,
                             const float* __restrict__ vw2,
                             const float* __restrict__ vb2,
                             float* __restrict__ out, int G, int Kd)
{
    int g = blockIdx.x * blockDim.x + threadIdx.x;
    if (g < G) {
        float s = vb2[0];
        for (int k = 0; k < Kd; ++k) s += v1[(size_t)g * Kd + k] * vw2[k];
        out[g] = tanhf(s);
    }
}

// ---------------------------------------------------------------------------
// Host launcher
// ---------------------------------------------------------------------------
static inline int ceil_div(int a, int b) { return (a + b - 1) / b; }
static inline int padded_tiles(int N) { return ceil_div(ceil_div(N, 16), 4) * 4; }
static inline size_t packed_elems(int K, int N) { return (size_t)padded_tiles(N) * K * 16; }

static void launch_gemm(bool relu, bool biasflag, bool outbf,
                        const __bf16* A, const __bf16* Bp, const float* bias, void* C,
                        int M, int N, int K, hipStream_t stream)
{
    int nGroups = padded_tiles(N) / 4;
    dim3 block(32, 4);
    dim3 grid(M / 16, ceil_div(nGroups, 4));
#define LG(R, B, O) wmma_gemm_kernel<R, B, O><<<grid, block, 0, stream>>>(A, Bp, bias, C, M, N, K, nGroups)
    if (relu) {
        if (biasflag) { if (outbf) LG(true, true, true);  else LG(true, true, false); }
        else          { if (outbf) LG(true, false, true); else LG(true, false, false); }
    } else {
        if (biasflag) { if (outbf) LG(false, true, true);  else LG(false, true, false); }
        else          { if (outbf) LG(false, false, true); else LG(false, false, false); }
    }
#undef LG
}

extern "C" void kernel_launch(void* const* d_in, const int* in_sizes, int n_in,
                              void* d_out, int out_size, void* d_ws, size_t ws_size,
                              hipStream_t stream)
{
    // ---- inputs (setup_inputs order) ----
    const float* x    = (const float*)d_in[0];
    const int*   ei   = (const int*)  d_in[1];
    const int*   batch= (const int*)  d_in[2];
    const float* W0   = (const float*)d_in[3];
    const float* b0   = (const float*)d_in[4];
    const float* Wr   = (const float*)d_in[5];
    const float* br   = (const float*)d_in[6];
    const float* pw1  = (const float*)d_in[7];
    const float* pb1  = (const float*)d_in[8];
    const float* pw2  = (const float*)d_in[9];
    const float* pb2  = (const float*)d_in[10];
    const float* vw1  = (const float*)d_in[11];
    const float* vb1  = (const float*)d_in[12];
    const float* vw2  = (const float*)d_in[13];
    const float* vb2  = (const float*)d_in[14];

    const int F   = 128;
    const int NND = in_sizes[0] / F;             // 50000 (multiple of 16)
    const int E   = in_sizes[1] / 2;             // 800000
    const int H   = in_sizes[4];                 // 256
    const int HH  = in_sizes[8];                 // 128
    const int P   = in_sizes[10];                // 140
    const int L1  = in_sizes[5] / (H * H);       // 3
    const int G   = out_size / (P + 1);          // 512
    (void)n_in; (void)ws_size;

    const int* src = ei;
    const int* dst = ei + E;

    // ---- workspace carve (64B aligned regions) ----
    char* base = (char*)d_ws;
    size_t off = 0;
    auto carve = [&](size_t bytes) -> void* {
        void* r = base + off;
        off += (bytes + 63) & ~(size_t)63;
        return r;
    };
    __bf16* xbf   = (__bf16*)carve((size_t)NND * F * 2);
    __bf16* hbfA  = (__bf16*)carve((size_t)NND * H * 2);
    __bf16* hbfB  = (__bf16*)carve((size_t)NND * H * 2);
    float*  hw    = (float*) carve((size_t)NND * H * 4);
    float*  agg   = (float*) carve((size_t)NND * H * 4);
    float*  deg   = (float*) carve((size_t)NND * 4);
    float*  dis   = (float*) carve((size_t)NND * 4);
    float*  sums  = (float*) carve((size_t)G * H * 4);
    float*  cnts  = (float*) carve((size_t)G * 4);
    __bf16* gpool = (__bf16*)carve((size_t)G * H * 2);
    __bf16* p1    = (__bf16*)carve((size_t)G * HH * 2);
    float*  logits= (float*) carve((size_t)G * P * 4);
    float*  v1    = (float*) carve((size_t)G * HH * 4);
    __bf16* W0p   = (__bf16*)carve(packed_elems(F, H) * 2);
    __bf16* Wrp0  = (__bf16*)carve(packed_elems(H, H) * 2);
    __bf16* Wrp1  = (__bf16*)carve(packed_elems(H, H) * 2);
    __bf16* Wrp2  = (__bf16*)carve(packed_elems(H, H) * 2);
    __bf16* pw1p  = (__bf16*)carve(packed_elems(H, HH) * 2);
    __bf16* pw2p  = (__bf16*)carve(packed_elems(HH, P) * 2);
    __bf16* vw1p  = (__bf16*)carve(packed_elems(H, HH) * 2);
    __bf16* WrpAll[3] = {Wrp0, Wrp1, Wrp2};

    float* out_policy = (float*)d_out;                      // [G, P]
    float* out_value  = (float*)d_out + (size_t)G * P;      // [G]

    // ---- pack weights into WMMA fragment order (bf16, group-interleaved) ----
    auto packW = [&](const float* W, __bf16* Bp, int K, int N) {
        int nTilesPad = padded_tiles(N);
        int total = nTilesPad * (K >> 5) * 32;
        pack_weight_kernel<<<ceil_div(total, 256), 256, 0, stream>>>(W, Bp, K, N, nTilesPad);
    };
    packW(W0, W0p, F, H);
    for (int i = 0; i < L1 && i < 3; ++i)
        packW(Wr + (size_t)i * H * H, WrpAll[i], H, H);
    packW(pw1, pw1p, H, HH);
    packW(pw2, pw2p, HH, P);
    packW(vw1, vw1p, H, HH);

    // ---- x -> bf16 ----
    {
        long long n = (long long)NND * F;
        f32_to_bf16_kernel<<<(int)((n + 255) / 256), 256, 0, stream>>>(x, xbf, n);
    }

    // ---- degree & symmetric norm ----
    hipMemsetAsync(deg, 0, (size_t)NND * sizeof(float), stream);
    degree_kernel<<<ceil_div(E, 256), 256, 0, stream>>>(dst, deg, E);
    rsqrt_kernel<<<ceil_div(NND, 256), 256, 0, stream>>>(deg, dis, NND);

    const long long nodeElems = (long long)NND * H;
    const int seBlocks = (int)((nodeElems + 255) / 256);

    // ---- GCN layers: hw = A@W (WMMA) ; agg = scatter ; h_next = relu(...) ----
    const __bf16* hin = xbf;
    __bf16* hout = hbfA;
    int Kcur = F;
    const __bf16* Wp = W0p;
    const float* bcur = b0;
    for (int layer = 0; layer <= L1; ++layer) {
        launch_gemm(false, false, false, hin, Wp, nullptr, hw, NND, H, Kcur, stream);
        hipMemsetAsync(agg, 0, (size_t)NND * H * sizeof(float), stream);
        edge_aggregate_kernel<<<4096, 256, 0, stream>>>(hw, src, dst, dis, agg, E, H);
        selfloop_bias_relu_kernel<<<seBlocks, 256, 0, stream>>>(agg, hw, dis, bcur, hout, NND, H);
        hin  = hout;
        hout = (hout == hbfA) ? hbfB : hbfA;
        Kcur = H;
        if (layer < L1) {
            Wp   = WrpAll[layer < 3 ? layer : 2];
            bcur = br + (size_t)layer * H;
        }
    }
    const __bf16* hFinal = hin;   // last written buffer

    // ---- global mean pool ----
    hipMemsetAsync(sums, 0, (size_t)G * H * sizeof(float), stream);
    hipMemsetAsync(cnts, 0, (size_t)G * sizeof(float), stream);
    {
        long long total = (long long)NND * (H >> 2);
        pool_accum_kernel<<<(int)((total + 255) / 256), 256, 0, stream>>>(hFinal, batch, sums, cnts, NND, H);
    }
    pool_div_kernel<<<ceil_div(G * H, 256), 256, 0, stream>>>(sums, cnts, gpool, G, H);

    // ---- policy head ----
    launch_gemm(true,  true, true,  gpool, pw1p, pb1, p1,     G, HH, H,  stream);
    launch_gemm(false, true, false, p1,    pw2p, pb2, logits, G, P,  HH, stream);
    softmax_kernel<<<G, 256, 0, stream>>>(logits, out_policy, P);

    // ---- value head ----
    launch_gemm(true, true, false, gpool, vw1p, vb1, v1, G, HH, H, stream);
    value_kernel<<<ceil_div(G, 256), 256, 0, stream>>>(v1, vw2, vb2, out_value, G, HH);
}
